// NeuralODE_82257213653102
// MI455X (gfx1250) — compile-verified, matrix operand order checked by reading, not run on hardware
//
#include <hip/hip_runtime.h>
#include <math.h>

// ---------------------------------------------------------------------------
// Neural ODE (RK4, 20 steps) for MI455X / gfx1250.
//
// - bf16 WMMA (v_wmma_f32_16x16x32_bf16) with fp32 accumulation; fp32 ODE
//   state and RK4 combination. Working set (~34 MB) is L2-resident (192 MB).
// - Block-level LDS staging via CDNA5 async copies
//   (global_load_async_to_lds_b128, ASYNCcnt) with double buffering:
//   cuts L2 operand traffic ~3x vs per-wave global fragment loads
//   (~128 MB vs ~384 MB per 2048^2 GEMM -> ~7 TB/s demand, comfortably
//   within the memory system at matrix-core speed).
// - The time-column concat is folded into layer-1 bias: b1 + t * W1[1024,:].
// ---------------------------------------------------------------------------

typedef __attribute__((ext_vector_type(16))) __bf16       v16bf;
typedef __attribute__((ext_vector_type(8)))  float        v8f;
typedef __attribute__((ext_vector_type(4)))  unsigned int v4u;

union FragAB {
    v16bf v;
    v4u   u[2];
};

#define ODE_STEPS 20

// One 16-byte async copy, global -> LDS, per active lane (ASYNCcnt-tracked).
__device__ __forceinline__ void async_cp16(unsigned lds_byte_off, const __bf16* g)
{
    asm volatile("global_load_async_to_lds_b128 %0, %1, off"
                 :: "v"(lds_byte_off),
                    "v"((unsigned long long)(size_t)g)
                 : "memory");
}

__device__ __forceinline__ void wait_async0()
{
    asm volatile("s_wait_asynccnt 0x0" ::: "memory");
}

// Low 32 bits of a flat LDS pointer == LDS byte address (ISA: aperture maps
// addr[31:0] to the workgroup's LDS offset).
__device__ __forceinline__ unsigned lds_off(const void* p)
{
    return (unsigned)(size_t)p;
}

// ---------------------------------------------------------------------------
// GEMM: C[M,N] = act( A[M,K](bf16) * Bt[N,K]^T(bf16) + bias (+ t*tbias) )
//
// Block = 256 threads (8 waves), tile 128(M) x 128(N); wave tile 64x32
// (4x2 WMMA tiles). K-chunk = 32. A/B 128x32 slabs are staged in LDS by
// async b128 copies (4 per wave per chunk), double-buffered with one
// s_wait_asynccnt(0) + one barrier per chunk.
//
// LDS row stride = 40 bf16 (80 B): fragment ds_load_b128 reads advance
// 20 banks per row -> conflict-free.
//
// Fragment layouts (ISA 7.12.2, wave32, 16-bit):
//  A 16x32: lane&15 = row; lanes<16 hold K=0..7 & 16..23, lanes>=16 K=8..15 &
//  24..31 -> two contiguous 16B reads at khalf and khalf+16.
//  B identical over columns because Bt is stored [N,K].
//  C/D: vgpr r -> row r + 8*(lane>>4); col = lane&15.
// ---------------------------------------------------------------------------
template <bool TANH, bool BF16OUT, bool TBIAS>
__global__ __launch_bounds__(256) void gemm_wmma_bf16(
    const __bf16* __restrict__ A,     // [M,K] row-major bf16
    const __bf16* __restrict__ Bt,    // [N,K] row-major bf16 (i.e. B^T)
    const float*  __restrict__ bias,  // [N]
    const float*  __restrict__ tbias, // [N] (only if TBIAS)
    float t,
    void* __restrict__ Cout,          // [M,N] bf16 or f32
    int M, int N, int K)
{
    constexpr int KC  = 32;   // K per chunk
    constexpr int LST = 40;   // LDS row stride in bf16 elements (80 B)

    __shared__ __bf16 sA[2][128 * LST];
    __shared__ __bf16 sB[2][128 * LST];

    const int tid   = threadIdx.x;
    const int lane  = tid & 31;
    const int wave  = tid >> 5;
    const int waveM = wave >> 2;                  // 0..1
    const int waveN = wave & 3;                   // 0..3
    const int l15   = lane & 15;
    const int rhalf = lane >> 4;                  // 0 or 1
    const int khalf = rhalf * 8;

    const int mBlk = blockIdx.y * 128;
    const int nBlk = blockIdx.x * 128;

    // ---- async copy plan: slab = 128 rows x 32 elems = 512 x 16B units;
    //      256 threads x 2 rounds; unit u: row = u>>2, seg = u&3.
    const int uA0 = tid, uA1 = tid + 256;
    const __bf16* gA0 = A  + (size_t)(mBlk + (uA0 >> 2)) * (size_t)K + (uA0 & 3) * 8;
    const __bf16* gA1 = A  + (size_t)(mBlk + (uA1 >> 2)) * (size_t)K + (uA1 & 3) * 8;
    const __bf16* gB0 = Bt + (size_t)(nBlk + (uA0 >> 2)) * (size_t)K + (uA0 & 3) * 8;
    const __bf16* gB1 = Bt + (size_t)(nBlk + (uA1 >> 2)) * (size_t)K + (uA1 & 3) * 8;

    unsigned dA0[2], dA1[2], dB0[2], dB1[2];
#pragma unroll
    for (int b = 0; b < 2; ++b) {
        dA0[b] = lds_off(&sA[b][(uA0 >> 2) * LST + (uA0 & 3) * 8]);
        dA1[b] = lds_off(&sA[b][(uA1 >> 2) * LST + (uA1 & 3) * 8]);
        dB0[b] = lds_off(&sB[b][(uA0 >> 2) * LST + (uA0 & 3) * 8]);
        dB1[b] = lds_off(&sB[b][(uA1 >> 2) * LST + (uA1 & 3) * 8]);
    }

    v8f acc[4][2];
#pragma unroll
    for (int i = 0; i < 4; ++i)
#pragma unroll
        for (int j = 0; j < 2; ++j)
            acc[i][j] = (v8f)0.0f;

    const int numChunks = K / KC;

    // preload chunk 0 into buffer 0 (4 async ops per wave)
    async_cp16(dA0[0], gA0);
    async_cp16(dA1[0], gA1);
    async_cp16(dB0[0], gB0);
    async_cp16(dB1[0], gB1);

    for (int c = 0; c < numChunks; ++c) {
        const int cur = c & 1;

        // my copies for chunk c are done...
        wait_async0();
        // ...and after this barrier everyone's are, and everyone has finished
        // reading buffer (1-cur) [consumed in iteration c-1], so it is safe
        // to overwrite it with chunk c+1.
        __syncthreads();

        if (c + 1 < numChunks) {
            const int k1 = (c + 1) * KC;
            async_cp16(dA0[1 - cur], gA0 + k1);
            async_cp16(dA1[1 - cur], gA1 + k1);
            async_cp16(dB0[1 - cur], gB0 + k1);
            async_cp16(dB1[1 - cur], gB1 + k1);
        }

        // ---- fragments from LDS
        FragAB a[4];
        FragAB b[2];
#pragma unroll
        for (int i = 0; i < 4; ++i) {
            const __bf16* p = &sA[cur][(waveM * 64 + i * 16 + l15) * LST + khalf];
            a[i].u[0] = *(const v4u*)(p);        // K = kb .. kb+7
            a[i].u[1] = *(const v4u*)(p + 16);   // K = kb+16 .. kb+23
        }
#pragma unroll
        for (int j = 0; j < 2; ++j) {
            const __bf16* p = &sB[cur][(waveN * 32 + j * 16 + l15) * LST + khalf];
            b[j].u[0] = *(const v4u*)(p);
            b[j].u[1] = *(const v4u*)(p + 16);
        }
#pragma unroll
        for (int i = 0; i < 4; ++i)
#pragma unroll
            for (int j = 0; j < 2; ++j)
                acc[i][j] = __builtin_amdgcn_wmma_f32_16x16x32_bf16(
                    /*neg_a=*/false, a[i].v,
                    /*neg_b=*/false, b[j].v,
                    /*c_mod=*/(short)0, acc[i][j],
                    /*reuse_a=*/false, /*reuse_b=*/false);
    }

    // ---- epilogue: bias (+ t*tbias), optional tanh, store bf16 or f32
    const int mBase = mBlk + waveM * 64;
    const int nBase = nBlk + waveN * 32;

    float bv[2];
#pragma unroll
    for (int j = 0; j < 2; ++j) {
        const int col = nBase + j * 16 + l15;
        float v = bias[col];
        if (TBIAS) v += t * tbias[col];
        bv[j] = v;
    }

#pragma unroll
    for (int i = 0; i < 4; ++i) {
#pragma unroll
        for (int j = 0; j < 2; ++j) {
            const int col = nBase + j * 16 + l15;
#pragma unroll
            for (int r = 0; r < 8; ++r) {
                const int row = mBase + i * 16 + r + 8 * rhalf;
                float v = acc[i][j][r] + bv[j];
                if (TANH) v = tanhf(v);
                if (BF16OUT)
                    ((__bf16*)Cout)[(size_t)row * (size_t)N + col] = (__bf16)v;
                else
                    ((float*)Cout)[(size_t)row * (size_t)N + col] = v;
            }
        }
    }
}

// ---------------------------------------------------------------------------
// fp32 [K,N] -> bf16 transposed [N,K]  (one-shot weight prep, L2 resident)
// ---------------------------------------------------------------------------
__global__ __launch_bounds__(256) void cvt_transpose_bf16(
    const float* __restrict__ W, __bf16* __restrict__ Wt, int K, int N)
{
    const int idx = blockIdx.x * blockDim.x + threadIdx.x;
    if (idx >= K * N) return;
    const int kk = idx / N;
    const int nn = idx - kk * N;
    Wt[(size_t)nn * (size_t)K + kk] = (__bf16)W[idx];
}

// z = z0 (fp32 state), Abuf = bf16(z0)
__global__ __launch_bounds__(256) void ode_init(
    const float* __restrict__ z0, float* __restrict__ z,
    __bf16* __restrict__ Abuf, int n)
{
    const int i = blockIdx.x * blockDim.x + threadIdx.x;
    if (i >= n) return;
    const float v = z0[i];
    z[i]    = v;
    Abuf[i] = (__bf16)v;
}

// RK4 combination (fp32):
//  mode 0 (after k1): znew = z + c*k ; Abuf = bf16(z + a*k)
//  mode 1 (after k2/k3): znew += c*k ; Abuf = bf16(z + a*k)
//  mode 2 (after k4): z = znew + c*k ; Abuf = bf16(z)   (advance step)
__global__ __launch_bounds__(256) void rk4_combine(
    float* __restrict__ z, const float* __restrict__ k,
    float* __restrict__ znew, __bf16* __restrict__ Abuf,
    float a, float c, int mode, int n)
{
    const int i = blockIdx.x * blockDim.x + threadIdx.x;
    if (i >= n) return;
    const float kv = k[i];
    if (mode == 2) {
        const float zf = znew[i] + c * kv;
        z[i]    = zf;
        Abuf[i] = (__bf16)zf;
    } else {
        const float zi = z[i];
        const float zn = (mode == 0) ? (zi + c * kv) : (znew[i] + c * kv);
        znew[i] = zn;
        Abuf[i] = (__bf16)(zi + a * kv);
    }
}

// ---------------------------------------------------------------------------
extern "C" void kernel_launch(void* const* d_in, const int* in_sizes, int n_in,
                              void* d_out, int out_size, void* d_ws, size_t ws_size,
                              hipStream_t stream)
{
    (void)in_sizes; (void)n_in; (void)out_size; (void)ws_size;

    const float* z0 = (const float*)d_in[0];   // [1024,1024]
    const float* W1 = (const float*)d_in[1];   // [1025,2048]
    const float* b1 = (const float*)d_in[2];   // [2048]
    const float* W2 = (const float*)d_in[3];   // [2048,2048]
    const float* b2 = (const float*)d_in[4];   // [2048]
    const float* W3 = (const float*)d_in[5];   // [2048,1024]
    const float* b3 = (const float*)d_in[6];   // [1024]

    const int B = 1024, D = 1024, H = 2048;
    float* z = (float*)d_out;                  // fp32 ODE state lives in d_out

    // workspace carve-out (~34 MB)
    char* p = (char*)d_ws;
    auto carve = [&](size_t bytes) -> char* {
        char* q = p;
        p += (bytes + 255) & ~(size_t)255;
        return q;
    };
    __bf16* W1t  = (__bf16*)carve((size_t)H * D * 2);  // [H=2048, K=1024]
    __bf16* W2t  = (__bf16*)carve((size_t)H * H * 2);  // [H, H]
    __bf16* W3t  = (__bf16*)carve((size_t)D * H * 2);  // [D=1024, K=2048]
    __bf16* Abuf = (__bf16*)carve((size_t)B * D * 2);  // GEMM input, stage z
    __bf16* h1   = (__bf16*)carve((size_t)B * H * 2);
    __bf16* h2   = (__bf16*)carve((size_t)B * H * 2);
    float*  kbuf = (float*) carve((size_t)B * D * 4);
    float*  znew = (float*) carve((size_t)B * D * 4);

    const int EThreads = 256;
    // weight prep: first 1024 rows of W1 (the t-column row is folded into bias)
    cvt_transpose_bf16<<<(D * H + EThreads - 1) / EThreads, EThreads, 0, stream>>>(W1, W1t, D, H);
    cvt_transpose_bf16<<<(H * H + EThreads - 1) / EThreads, EThreads, 0, stream>>>(W2, W2t, H, H);
    cvt_transpose_bf16<<<(H * D + EThreads - 1) / EThreads, EThreads, 0, stream>>>(W3, W3t, H, D);
    ode_init<<<(B * D + EThreads - 1) / EThreads, EThreads, 0, stream>>>(z0, z, Abuf, B * D);

    const float* W1last = W1 + (size_t)D * H;  // row 1024: the time-column weights
    const float  hh = 1.0f / (float)ODE_STEPS;

    const dim3 gH(H / 128, B / 128);   // layers 1 & 2: N = 2048
    const dim3 gD(D / 128, B / 128);   // layer 3:      N = 1024
    const int  nBD = B * D;
    const int  cBlocks = (nBD + EThreads - 1) / EThreads;

    auto mlp = [&](float t) {
        gemm_wmma_bf16<true,  true,  true ><<<gH, 256, 0, stream>>>(
            Abuf, W1t, b1, W1last, t, (void*)h1, B, H, D);
        gemm_wmma_bf16<true,  true,  false><<<gH, 256, 0, stream>>>(
            h1, W2t, b2, nullptr, 0.0f, (void*)h2, B, H, H);
        gemm_wmma_bf16<false, false, false><<<gD, 256, 0, stream>>>(
            h2, W3t, b3, nullptr, 0.0f, (void*)kbuf, B, D, H);
    };

    for (int s = 0; s < ODE_STEPS; ++s) {
        const float t = hh * (float)s;
        mlp(t);                                                    // k1
        rk4_combine<<<cBlocks, EThreads, 0, stream>>>(z, kbuf, znew, Abuf,
                                                      0.5f * hh, hh / 6.0f, 0, nBD);
        mlp(t + 0.5f * hh);                                        // k2
        rk4_combine<<<cBlocks, EThreads, 0, stream>>>(z, kbuf, znew, Abuf,
                                                      0.5f * hh, hh / 3.0f, 1, nBD);
        mlp(t + 0.5f * hh);                                        // k3
        rk4_combine<<<cBlocks, EThreads, 0, stream>>>(z, kbuf, znew, Abuf,
                                                      hh, hh / 3.0f, 1, nBD);
        mlp(t + hh);                                               // k4
        rk4_combine<<<cBlocks, EThreads, 0, stream>>>(z, kbuf, znew, Abuf,
                                                      0.0f, hh / 6.0f, 2, nBD);
    }
}
